// NeighborhoodAttentionAggregator_6923487281993
// MI455X (gfx1250) — matrix operand order repaired and checked by reading, not compiled, not run on hardware
//
#include <hip/hip_runtime.h>
#include <stdint.h>

// ---------------------------------------------------------------------------
// CDNA5 (gfx1250) neighborhood-attention aggregator.
// bf16 WMMA (v_wmma_f32_16x16x32_bf16) for Q/K/V/Wo GEMMs, f32 accumulate.
// 16 wave32 per block, one row per wave, 16 rows per block.
// ---------------------------------------------------------------------------

typedef __attribute__((ext_vector_type(16))) __bf16 v16bf;
typedef __attribute__((ext_vector_type(8)))  float  v8f;

union Frag {
    v16bf    bf;
    uint32_t u[8];
};

#define D_EMB   256
#define TA      128
#define KNBR    16
#define ROWS    16   // rows per block == waves per block

// Build a 16x32 bf16 A-fragment element set for one lane from row-major f32.
// Per ISA 7.12.2 (16-bit A 16x32): lane half h: e0..7 -> K = base+8h+e,
// e8..15 -> K = base+8h+16+(e-8). Caller passes p = row + 32*c + 8*half.
// Native f32->bf16 converts (RNE fptrunc -> v_cvt_pk_bf16_f32 class).
__device__ inline void build_frag_f32(Frag& f, const float* p) {
    float v[16];
    *(float4*)(v + 0)  = *(const float4*)(p);
    *(float4*)(v + 4)  = *(const float4*)(p + 4);
    *(float4*)(v + 8)  = *(const float4*)(p + 16);
    *(float4*)(v + 12) = *(const float4*)(p + 20);
#pragma unroll
    for (int e = 0; e < 16; ++e) f.bf[e] = (__bf16)v[e];
}

// Load a pre-swizzled B-fragment: 32 bytes/lane, contiguous in workspace.
__device__ inline void load_frag_b(Frag& f, const unsigned short* base,
                                   int fragIdx, int lane) {
    const uint4* p = (const uint4*)(base + ((size_t)fragIdx * 32 + lane) * 16);
    uint4 x = p[0];
    uint4 y = p[1];
    f.u[0] = x.x; f.u[1] = x.y; f.u[2] = x.z; f.u[3] = x.w;
    f.u[4] = y.x; f.u[5] = y.y; f.u[6] = y.z; f.u[7] = y.w;
}

__device__ inline v8f wmma_bf16(const Frag& a, const Frag& b, v8f c) {
    return __builtin_amdgcn_wmma_f32_16x16x32_bf16(
        /*neg_a=*/false, a.bf, /*neg_b=*/false, b.bf,
        /*c_mod=*/(short)0, c, /*reuse_a=*/false, /*reuse_b=*/false);
}

// ---------------------------------------------------------------------------
// Prep: convert + swizzle Wq/Wk/Wv (256x128) and Wo (128x256) f32 -> bf16
// B-fragments.  B-fragment layout (32x16 bf16 B operand): lane half h,
// element e -> K = 32*c + 8*h + e (e<8), 32*c + 16 + 8*h + (e-8) (e>=8) ; N = 16*nt + (lane&15).
// ws layout: [w][frag=nt*nk+c][lane][e], 64 frags * 512 bf16 per weight.
// ---------------------------------------------------------------------------
__global__ void nat_prep(const float* __restrict__ Wq, const float* __restrict__ Wk,
                         const float* __restrict__ Wv, const float* __restrict__ Wo,
                         unsigned short* __restrict__ ws) {
    int gid = blockIdx.x * blockDim.x + threadIdx.x;
    if (gid >= 4 * 64 * 32 * 16) return;
    int e    = gid & 15;
    int lane = (gid >> 4) & 31;
    int f    = (gid >> 9) & 63;
    int w    = gid >> 15;
    int half = lane >> 4;
    int l15  = lane & 15;
    const float* W;
    int nk, Ndim;
    if (w < 3) { W = (w == 0 ? Wq : (w == 1 ? Wk : Wv)); nk = 8; Ndim = 128; }
    else       { W = Wo;                                  nk = 4; Ndim = 256; }
    int nt = f / nk;
    int c  = f % nk;
    int Kidx = c * 32 + (e < 8 ? 8 * half + e : 16 + 8 * half + (e - 8));
    int Nidx = nt * 16 + l15;
    __bf16 b = (__bf16)W[(size_t)Kidx * Ndim + Nidx];
    ws[gid] = __builtin_bit_cast(unsigned short, b);
}

// ---------------------------------------------------------------------------
// Main kernel: 512 threads = 16 wave32, 16 rows per block.
// ---------------------------------------------------------------------------
__launch_bounds__(512, 1)
__global__ void nat_main(const float* __restrict__ center,
                         const float* __restrict__ allemb,
                         const int*   __restrict__ nidx,
                         const float* __restrict__ nw,
                         const unsigned short* __restrict__ wq_f,
                         const unsigned short* __restrict__ wk_f,
                         const unsigned short* __restrict__ wv_f,
                         const unsigned short* __restrict__ wo_f,
                         const float* __restrict__ Wg,
                         const float* __restrict__ bg,
                         const float* __restrict__ gamma,
                         const float* __restrict__ beta,
                         float* __restrict__ out) {
    __shared__ float qbuf[ROWS][TA];      // Q rows for this block
    __shared__ float ctxbuf[ROWS][TA];    // attention context per row
    __shared__ float ctx2[ROWS][D_EMB];   // ctx @ Wo

    const int tid  = threadIdx.x;
    const int wave = tid >> 5;
    const int lane = tid & 31;
    const int l15  = lane & 15;
    const int half = lane >> 4;
    const int row0 = blockIdx.x * ROWS;
    const int row  = row0 + wave;

    // ---- Phase 0: Q = centers[16,256] @ Wq -> qbuf (waves 0..7, 1 N-tile ea)
    if (wave < 8) {
        const int nt = wave;
        v8f acc;
#pragma unroll
        for (int r = 0; r < 8; ++r) acc[r] = 0.0f;
        const float* arow = center + (size_t)(row0 + l15) * D_EMB;
#pragma unroll 2
        for (int c = 0; c < 8; ++c) {
            Frag a, b;
            build_frag_f32(a, arow + c * 32 + half * 8);
            load_frag_b(b, wq_f, nt * 8 + c, lane);
            acc = wmma_bf16(a, b, acc);
        }
#pragma unroll
        for (int r = 0; r < 8; ++r)
            qbuf[half * 8 + r][nt * 16 + l15] = acc[r];   // C layout: M=8*half+r
    }
    __syncthreads();

    // ---- Phase 1: per-row attention (one wave per row) -------------------
    // Gather pointer: A-matrix row M = lane&15 -> neighbor (row,M).
    const int    gidx = nidx[row * KNBR + l15];
    const float* nrow = allemb + (size_t)gidx * D_EMB;

    v8f ak[8], av[8];
#pragma unroll
    for (int nt = 0; nt < 8; ++nt) {
#pragma unroll
        for (int r = 0; r < 8; ++r) { ak[nt][r] = 0.0f; av[nt][r] = 0.0f; }
    }
#pragma unroll 2
    for (int c = 0; c < 8; ++c) {
        Frag a;
        build_frag_f32(a, nrow + c * 32 + half * 8);   // shared by K and V GEMMs
#pragma unroll
        for (int nt = 0; nt < 8; ++nt) {
            Frag bk, bv;
            load_frag_b(bk, wk_f, nt * 8 + c, lane);
            load_frag_b(bv, wv_f, nt * 8 + c, lane);
            ak[nt] = wmma_bf16(a, bk, ak[nt]);
            av[nt] = wmma_bf16(a, bv, av[nt]);
        }
    }

    // scores[h][M] with M = 8*half + r; q from LDS, partial over 16 lanes.
    float ql[8];
#pragma unroll
    for (int nt = 0; nt < 8; ++nt) ql[nt] = qbuf[wave][nt * 16 + l15];
    float nwv[8];
#pragma unroll
    for (int r = 0; r < 8; ++r) nwv[r] = nw[row * KNBR + half * 8 + r];

    const float scale = 0.125f;   // 64^-0.5
    float sc[2][8];
#pragma unroll
    for (int h = 0; h < 2; ++h) {
#pragma unroll
        for (int r = 0; r < 8; ++r) {
            float p = 0.0f;
#pragma unroll
            for (int j = 0; j < 4; ++j) p += ql[h * 4 + j] * ak[h * 4 + j][r];
            p += __shfl_xor(p, 1);
            p += __shfl_xor(p, 2);
            p += __shfl_xor(p, 4);
            p += __shfl_xor(p, 8);   // xor<16: reduction stays within the half
            float w = fmaxf(nwv[r], 1e-6f);
            sc[h][r] = p * scale + __logf(w + 1e-6f);
        }
    }

    // softmax over 16 keys per head (exchange halves with xor 16)
    float attn[2][8];
#pragma unroll
    for (int h = 0; h < 2; ++h) {
        float oth[8];
#pragma unroll
        for (int r = 0; r < 8; ++r) oth[r] = __shfl_xor(sc[h][r], 16);
        float mx = -1e30f;
#pragma unroll
        for (int r = 0; r < 8; ++r) mx = fmaxf(mx, fmaxf(sc[h][r], oth[r]));
        float s = 0.0f;
#pragma unroll
        for (int r = 0; r < 8; ++r) {
            attn[h][r] = __expf(sc[h][r] - mx);
            s += attn[h][r] + __expf(oth[r] - mx);
        }
        float inv = 1.0f / s;
#pragma unroll
        for (int r = 0; r < 8; ++r) attn[h][r] *= inv;
    }

    // ctx[col] = sum_M attn[h][M] * v[M][col]; half-partial + xor16 combine
#pragma unroll
    for (int nt = 0; nt < 8; ++nt) {
        const int h = nt >> 2;
        float p = 0.0f;
#pragma unroll
        for (int r = 0; r < 8; ++r) p += attn[h][r] * av[nt][r];
        p += __shfl_xor(p, 16);
        if (half == 0) ctxbuf[wave][nt * 16 + l15] = p;
    }
    __syncthreads();

    // ---- Phase 2: ctx[16,128] @ Wo[128,256]; wave w owns N-tile w ---------
    {
        v8f acc;
#pragma unroll
        for (int r = 0; r < 8; ++r) acc[r] = 0.0f;
        const float* crow = &ctxbuf[l15][0];   // A row M = lane&15 (LDS)
#pragma unroll 2
        for (int c = 0; c < 4; ++c) {
            Frag a, b;
            build_frag_f32(a, crow + c * 32 + half * 8);
            load_frag_b(b, wo_f, wave * 4 + c, lane);
            acc = wmma_bf16(a, b, acc);
        }
#pragma unroll
        for (int r = 0; r < 8; ++r)
            ctx2[half * 8 + r][wave * 16 + l15] = acc[r];
    }
    __syncthreads();

    // ---- Phase 3: gate + LayerNorm (one wave per row, 8 cols per lane) ----
    {
        const float* crow = center + (size_t)row * D_EMB;
        const int colb = lane * 8;
        float c8[8], x8[8];
#pragma unroll
        for (int j = 0; j < 8; ++j) {
            c8[j] = crow[colb + j];
            x8[j] = ctx2[wave][colb + j];
        }
        float g = 0.0f;
#pragma unroll
        for (int j = 0; j < 8; ++j)
            g += c8[j] * Wg[colb + j] + x8[j] * Wg[256 + colb + j];
#pragma unroll
        for (int m = 1; m < 32; m <<= 1) g += __shfl_xor(g, m);
        g = 1.0f / (1.0f + __expf(-(g + bg[0])));

        float agg[8], s = 0.0f, s2 = 0.0f;
#pragma unroll
        for (int j = 0; j < 8; ++j) {
            agg[j] = g * c8[j] + (1.0f - g) * x8[j];
            s  += agg[j];
            s2 += agg[j] * agg[j];
        }
#pragma unroll
        for (int m = 1; m < 32; m <<= 1) {
            s  += __shfl_xor(s,  m);
            s2 += __shfl_xor(s2, m);
        }
        const float mu  = s * (1.0f / 256.0f);
        const float var = s2 * (1.0f / 256.0f) - mu * mu;
        const float inv = rsqrtf(var + 1e-5f);
        float* orow = out + (size_t)row * D_EMB;
#pragma unroll
        for (int j = 0; j < 8; ++j)
            orow[colb + j] = (agg[j] - mu) * inv * gamma[colb + j] + beta[colb + j];
    }
}

// ---------------------------------------------------------------------------
extern "C" void kernel_launch(void* const* d_in, const int* in_sizes, int n_in,
                              void* d_out, int out_size, void* d_ws, size_t ws_size,
                              hipStream_t stream) {
    const float* center = (const float*)d_in[0];
    const float* allemb = (const float*)d_in[1];
    const int*   nidx   = (const int*)  d_in[2];
    const float* nw     = (const float*)d_in[3];
    const float* Wq     = (const float*)d_in[4];
    const float* Wk     = (const float*)d_in[5];
    const float* Wv     = (const float*)d_in[6];
    const float* Wo     = (const float*)d_in[7];
    const float* Wg     = (const float*)d_in[8];
    const float* bg     = (const float*)d_in[9];
    const float* gamma  = (const float*)d_in[10];
    const float* beta   = (const float*)d_in[11];

    unsigned short* wfrag = (unsigned short*)d_ws;   // 4 * 64KB of bf16 fragments

    // Swizzle weights into WMMA B-fragment layout (cheap, deterministic).
    nat_prep<<<(4 * 64 * 32 * 16 + 255) / 256, 256, 0, stream>>>(Wq, Wk, Wv, Wo, wfrag);

    const int B = in_sizes[0] / D_EMB;   // 32768
    nat_main<<<B / ROWS, 512, 0, stream>>>(
        center, allemb, nidx, nw,
        wfrag, wfrag + 32768, wfrag + 65536, wfrag + 98304,
        Wg, bg, gamma, beta, (float*)d_out);
}
